// GRUAutoEncoder_61607010893943
// MI455X (gfx1250) — matrix operand order repaired and checked by reading, not compiled, not run on hardware
//
#include <hip/hip_runtime.h>
#include <hip/hip_bf16.h>
#include <math.h>

// ---------------------------------------------------------------------------
// GRU auto-encoder on MI455X (gfx1250), wave32 + WMMA f32_16x16x32_f16,
// TDM (tensor_load_to_lds) for the per-step hidden-state broadcast.
// ---------------------------------------------------------------------------

typedef _Float16 half_t;
typedef __attribute__((ext_vector_type(16))) _Float16 v16h;
typedef __attribute__((ext_vector_type(8)))  _Float16 v8h;
typedef __attribute__((ext_vector_type(8)))  float    v8f;
typedef __attribute__((ext_vector_type(4)))  unsigned int u32x4;
typedef __attribute__((ext_vector_type(8)))  int i32x8;
typedef __attribute__((ext_vector_type(4)))  int i32x4;

#define DIM_T 1024
#define DIM_B 64
#define DIM_IN 64
#define DIM_H 512
#define DIM_E 128
#define DIM_G (3 * DIM_H)      // 1536
#define NROWS ((long long)DIM_T * DIM_B)   // 65536
#define REC_WGS 4              // workgroups cooperating on one recurrence

// LDS h matrix: 64 rows x 512 f16, row pitch padded to 520 f16 (1040 B =
// 260 banks == 4 mod 64 -> per-row bank rotation, conflict-free A fragments).
// The pad matches TDM pad_enable: pad_interval=7 (256 dwords), pad_amount=3
// (4 dwords) -> 16 bytes of pad per 1024-byte row.
#define H_PITCH 520
#define REC_LDS_BYTES (DIM_B * H_PITCH * 2)   // 66560

#if defined(__has_builtin)
#if __has_builtin(__builtin_amdgcn_tensor_load_to_lds) && \
    __has_builtin(__builtin_amdgcn_s_wait_tensorcnt)
#define USE_TDM 1
#endif
#endif

__device__ __forceinline__ v16h cat8(v8h lo, v8h hi) {
  return __builtin_shufflevector(lo, hi, 0, 1, 2, 3, 4, 5, 6, 7,
                                 8, 9, 10, 11, 12, 13, 14, 15);
}

// D = A*B + C, 16x16x32 f16 -> f32 accum
__device__ __forceinline__ v8f wmma_f16(v16h a, v16h b, v8f c) {
  return __builtin_amdgcn_wmma_f32_16x16x32_f16(false, a, false, b,
                                                (short)0, c, false, false);
}

// A fragment (16x32, M x K) from row-major f16 [*, lda].
// ISA layout: lanes 0-15 row M=lane hold K {k0..k0+7, k0+16..k0+23};
//             lanes 16-31 row M=lane-16 hold K {k0+8..15, k0+24..31}.
__device__ __forceinline__ v16h load_a_rowmajor(const half_t* __restrict__ base,
                                                int lda, int row0, int k0,
                                                int lane) {
  int r = row0 + (lane & 15);
  int ko = k0 + ((lane & 16) ? 8 : 0);
  const half_t* p = base + (size_t)r * lda + ko;
  v8h lo = *(const v8h*)(p);
  v8h hi = *(const v8h*)(p + 16);
  return cat8(lo, hi);
}

// B fragment (32x16, K x N) where B[k][n] = W[g0+n][k], W row-major [M,K].
// ISA layout: lanes 0-15 col N=lane hold K {k0..k0+7, k0+16..k0+23};
//             lanes 16-31 hold K {k0+8..15, k0+24..31}.
__device__ __forceinline__ v16h load_b_wt(const half_t* __restrict__ W, int K,
                                          int g0, int k0, int lane) {
  int g = g0 + (lane & 15);
  int ko = k0 + ((lane & 16) ? 8 : 0);
  const half_t* p = W + (size_t)g * K + ko;
  v8h lo = *(const v8h*)(p);
  v8h hi = *(const v8h*)(p + 16);
  return cat8(lo, hi);
}

// A fragment from padded LDS h [64][H_PITCH].
__device__ __forceinline__ v16h load_a_lds(const half_t* hs, int bt, int kc,
                                           int lane) {
  int row = (bt << 4) + (lane & 15);
  const half_t* rp = hs + row * H_PITCH;
  int off = (kc << 5) + ((lane & 16) ? 8 : 0);
  v8h lo = *(const v8h*)(rp + off);
  v8h hi = *(const v8h*)(rp + off + 16);
  return cat8(lo, hi);
}

__device__ __forceinline__ float sigmoid_f(float x) {
  return 1.0f / (1.0f + __expf(-x));
}

// ---------------------------------------------------------------------------
// Utility kernels
// ---------------------------------------------------------------------------
__global__ void f32_to_f16_kernel(const float* __restrict__ in,
                                  half_t* __restrict__ out, int n) {
  int i = blockIdx.x * blockDim.x + threadIdx.x;
  if (i < n) out[i] = (half_t)in[i];
}

__global__ void zero_u32_kernel(unsigned int* p, int n) {
  int i = blockIdx.x * blockDim.x + threadIdx.x;
  if (i < n) p[i] = 0u;
}

// ---------------------------------------------------------------------------
// Time-parallel GEMM: out[r, m] = sum_k A16[r,k] * W16[m,k] + bias[m]
// A16 row-major [Nrows, K] f16, W16 row-major [M, K] f16.
// Each wave computes a 16(row) x 64(col) block; 8 waves / 256-thread block.
// ---------------------------------------------------------------------------
__global__ void __launch_bounds__(256)
gemm_xw_kernel(const half_t* __restrict__ A, const half_t* __restrict__ W,
               const float* __restrict__ bias, half_t* __restrict__ out16,
               float* __restrict__ out32, int Nrows, int M, int K) {
  const int lane = threadIdx.x & 31;
  const int wave = blockIdx.x * (blockDim.x >> 5) + (threadIdx.x >> 5);
  const int colGroups = M >> 6;
  const int rowTiles = Nrows >> 4;
  if (wave >= rowTiles * colGroups) return;
  const int rt = wave / colGroups;
  const int cg = wave % colGroups;
  const int row0 = rt << 4;

  v8f acc[4] = {v8f{}, v8f{}, v8f{}, v8f{}};
  for (int k0 = 0; k0 < K; k0 += 32) {
    v16h a = load_a_rowmajor(A, K, row0, k0, lane);
#pragma unroll
    for (int c = 0; c < 4; ++c) {
      v16h b = load_b_wt(W, K, (cg << 6) + (c << 4), k0, lane);
      acc[c] = wmma_f16(a, b, acc[c]);
    }
  }
  // D layout: element i of v8f -> row = row0 + i + (lane>=16 ? 8 : 0),
  //           col = colbase + (lane & 15)
  const int radd = (lane & 16) ? 8 : 0;
  const int coff = lane & 15;
#pragma unroll
  for (int c = 0; c < 4; ++c) {
    const int col = (cg << 6) + (c << 4) + coff;
    const float bv = bias ? bias[col] : 0.0f;
#pragma unroll
    for (int i = 0; i < 8; ++i) {
      const float v = acc[c][i] + bv;
      const size_t idx = (size_t)(row0 + radd + i) * (size_t)M + col;
      if (out16) out16[idx] = (half_t)v;
      else       out32[idx] = v;
    }
  }
}

// ---------------------------------------------------------------------------
// GRU recurrence. REC_WGS workgroups x 1024 threads (32 wave32s).
// Workgroup wg owns j-tiles [8*wg, 8*wg+8): each wave owns one (j-tile,
// b-tile) pair and its r/z/n gate tiles, so all gate math stays in registers.
// Hidden state h is replicated per WGP in padded LDS (f16) and exchanged per
// step through a parity-double-buffered global buffer guarded by a monotonic
// agent-scope counter (release/acquire). The broadcast refresh is a single
// TDM tensor_load_to_lds DMA per WGP (with hardware row padding), tracked by
// TENSORcnt; sync-load fallback if the builtin is unavailable.
// ---------------------------------------------------------------------------
__global__ void __launch_bounds__(1024)
gru_rec_kernel(const half_t* __restrict__ gx,   // [T,B,3H] f16 (bih folded)
               unsigned long long gx_tstride,   // elems per t (0 => broadcast)
               const half_t* __restrict__ Whh,  // [3H, H] f16
               const float* __restrict__ bhh,   // [3H] f32
               half_t* __restrict__ ys,         // [T,B,H] f16 out
               half_t* __restrict__ hbuf,       // [2][B][H] f16 exchange
               unsigned int* __restrict__ counter,
               int T) {
  constexpr int B = DIM_B, H = DIM_H, G = DIM_G;
  extern __shared__ half_t h16[];               // [B][H_PITCH], padded rows

  const int tid = threadIdx.x;
  const int lane = tid & 31;
  const int w = tid >> 5;                        // wave in WG: 0..31
  const int wg = blockIdx.x;                     // 0..REC_WGS-1
  const int jt = (wg << 3) + (w >> 2);           // j-tile 0..31
  const int bt = w & 3;                          // b-tile 0..3
  const int jcol = (jt << 4) + (lane & 15);      // this lane's hidden column
  const int radd = (lane & 16) ? 8 : 0;
  const int brow0 = (bt << 4) + radd;

  const float bhr = bhh[jcol];
  const float bhz = bhh[H + jcol];
  const float bhn = bhh[2 * H + jcol];

  // h = 0 (pads included)
  for (int i = tid; i < B * H_PITCH; i += 1024) h16[i] = (half_t)0.0f;
  float h[8];
#pragma unroll
  for (int i = 0; i < 8; ++i) h[i] = 0.0f;
  __syncthreads();

  for (int t = 0; t < T; ++t) {
    if (t > 0) {
      // Wait until every workgroup has published h_{t-1}.
      if (tid == 0) {
        const unsigned int target = (unsigned int)(REC_WGS * t);
        while (__hip_atomic_load(counter, __ATOMIC_ACQUIRE,
                                 __HIP_MEMORY_SCOPE_AGENT) < target) {
          __builtin_amdgcn_s_sleep(2);
        }
      }
      __syncthreads();
      // Refresh full h (all 4 slices) from hbuf[(t-1)&1] into padded LDS.
      const half_t* src = hbuf + (size_t)((t - 1) & 1) * B * H;
#ifdef USE_TDM
      if (w == 0) {
        // Tensor DMA descriptor (D#), CDNA5 ISA:
        //   2D tile: tile_dim0 = tensor_dim0 = stride0 = 256 dwords (one
        //   512-f16 row), tile_dim1 = tensor_dim1 = 64 rows, data_size=4B,
        //   pad 4 dwords every 256 dwords -> LDS pitch 260 dwords.
        const unsigned long long ga = (unsigned long long)(size_t)src;
        const unsigned int ldsoff =
            (unsigned int)(size_t)(__attribute__((address_space(3))) half_t*)
                h16;
        u32x4 g0;
        g0.x = 1u;                                   // count=1, no gather
        g0.y = ldsoff;                               // lds_addr
        g0.z = (unsigned int)(ga & 0xffffffffull);   // global_addr[31:0]
        g0.w = (unsigned int)((ga >> 32) & 0x1ffffffull)  // global_addr[56:32]
               | (2u << 30);                         // type=2 ("image")
        i32x8 g1;
        g1[0] = (2 << 16)        // data_size = 4B
              | (1 << 20)        // pad_enable
              | (7 << 22)        // pad_interval = 256 dwords
              | (3 << 25);       // pad_amount  = 4 dwords
        g1[1] = (256 << 16);     // tensor_dim0[15:0] (atomic_barrier_addr=0)
        g1[2] = (64 << 16);      // tensor_dim0[31:16]=0, tensor_dim1[15:0]=64
        g1[3] = (256 << 16);     // tensor_dim1 hi = 0, tile_dim0 = 256
        g1[4] = 64;              // tile_dim1 = 64, tile_dim2 = 0
        g1[5] = 256;             // tensor_dim0_stride[31:0] = 256
        g1[6] = 0;               // stride hi, tensor_dim1_stride (2D: unused)
        g1[7] = 0;
        i32x4 gz4 = {0, 0, 0, 0};                    // groups 2/3: 2D tensor
        i32x8 gz8 = {0, 0, 0, 0, 0, 0, 0, 0};        // unused extension group
        __builtin_amdgcn_tensor_load_to_lds(g0, g1, gz4, gz4, gz8, 0);
        __builtin_amdgcn_s_wait_tensorcnt(0);
      }
#else
      for (int c = tid; c < B * (H / 8); c += 1024) {
        const int row = c >> 6;
        const int cc = c & 63;
        v8h v = *(const v8h*)(src + (size_t)row * H + (size_t)cc * 8);
        *(v8h*)(h16 + (size_t)row * H_PITCH + (size_t)cc * 8) = v;
      }
#endif
    }
    __syncthreads();

    // gh tiles for this wave's (j-tile, b-tile): r, z, n gates.
    v8f cr = {}, cz = {}, cn = {};
    for (int kc = 0; kc < 16; ++kc) {
      v16h a = load_a_lds(h16, bt, kc, lane);
      cr = wmma_f16(a, load_b_wt(Whh, H, (jt << 4),          kc * 32, lane), cr);
      cz = wmma_f16(a, load_b_wt(Whh, H, H + (jt << 4),      kc * 32, lane), cz);
      cn = wmma_f16(a, load_b_wt(Whh, H, 2 * H + (jt << 4),  kc * 32, lane), cn);
    }
    __syncthreads();   // everyone done reading LDS h before we overwrite it

    // Gate math + state update (register-resident h slice).
    const half_t* gxt = gx + (size_t)t * (size_t)gx_tstride;
    const size_t tysbase = (size_t)t * B * H;
    const size_t hpub = (size_t)(t & 1) * B * H;
#pragma unroll
    for (int i = 0; i < 8; ++i) {
      const int b = brow0 + i;
      const size_t base = (size_t)b * G + jcol;
      const float xr = (float)gxt[base];
      const float xz = (float)gxt[base + H];
      const float xn = (float)gxt[base + 2 * H];
      const float r = sigmoid_f(xr + cr[i] + bhr);
      const float z = sigmoid_f(xz + cz[i] + bhz);
      const float n = tanhf(xn + r * (cn[i] + bhn));
      const float hv = (1.0f - z) * n + z * h[i];
      h[i] = hv;
      const half_t hh = (half_t)hv;
      // padded LDS for next step's A fragments
      h16[(size_t)b * H_PITCH + jcol] = hh;
      // publish for the other workgroups + output sequence
      hbuf[hpub + (size_t)b * H + jcol] = hh;
      ys[tysbase + (size_t)b * H + jcol] = hh;
    }

    __threadfence();   // make hbuf/ys writes visible at agent scope
    __syncthreads();
    if (tid == 0) {
      __hip_atomic_fetch_add(counter, 1u, __ATOMIC_RELEASE,
                             __HIP_MEMORY_SCOPE_AGENT);
    }
  }
}

// ---------------------------------------------------------------------------
// Host side
// ---------------------------------------------------------------------------
extern "C" void kernel_launch(void* const* d_in, const int* in_sizes, int n_in,
                              void* d_out, int out_size, void* d_ws,
                              size_t ws_size, hipStream_t stream) {
  (void)in_sizes; (void)n_in; (void)out_size;

  // Input order from setup_inputs():
  const float* x      = (const float*)d_in[0];
  const float* eWih0  = (const float*)d_in[1];
  const float* eWhh0  = (const float*)d_in[2];
  const float* eBih0  = (const float*)d_in[3];
  const float* eBhh0  = (const float*)d_in[4];
  const float* eWih1  = (const float*)d_in[5];
  const float* eWhh1  = (const float*)d_in[6];
  const float* eBih1  = (const float*)d_in[7];
  const float* eBhh1  = (const float*)d_in[8];
  const float* elW    = (const float*)d_in[9];
  const float* elB    = (const float*)d_in[10];
  const float* dWih0  = (const float*)d_in[11];
  const float* dWhh0  = (const float*)d_in[12];
  const float* dBih0  = (const float*)d_in[13];
  const float* dBhh0  = (const float*)d_in[14];
  const float* dWih1  = (const float*)d_in[15];
  const float* dWhh1  = (const float*)d_in[16];
  const float* dBih1  = (const float*)d_in[17];
  const float* dBhh1  = (const float*)d_in[18];
  const float* dlW    = (const float*)d_in[19];
  const float* dlB    = (const float*)d_in[20];
  float* out = (float*)d_out;

  // Workspace bump allocator (256B aligned)
  char* base = (char*)d_ws;
  size_t off = 0;
  auto alloc = [&](size_t bytes) -> void* {
    void* p = base + off;
    off = (off + bytes + 255) & ~(size_t)255;
    return p;
  };

  unsigned int* counters = (unsigned int*)alloc(4 * sizeof(unsigned int));
  half_t* x16   = (half_t*)alloc((size_t)NROWS * DIM_IN * 2);
  half_t* z16   = (half_t*)alloc((size_t)DIM_B * DIM_E * 2);
  half_t* eW0_16 = (half_t*)alloc((size_t)DIM_G * DIM_IN * 2);
  half_t* eU0_16 = (half_t*)alloc((size_t)DIM_G * DIM_H * 2);
  half_t* eW1_16 = (half_t*)alloc((size_t)DIM_G * DIM_H * 2);
  half_t* eU1_16 = (half_t*)alloc((size_t)DIM_G * DIM_H * 2);
  half_t* elW16  = (half_t*)alloc((size_t)DIM_E * DIM_H * 2);
  half_t* dW0_16 = (half_t*)alloc((size_t)DIM_G * DIM_E * 2);
  half_t* dU0_16 = (half_t*)alloc((size_t)DIM_G * DIM_H * 2);
  half_t* dW1_16 = (half_t*)alloc((size_t)DIM_G * DIM_H * 2);
  half_t* dU1_16 = (half_t*)alloc((size_t)DIM_G * DIM_H * 2);
  half_t* dlW16  = (half_t*)alloc((size_t)DIM_IN * DIM_H * 2);
  half_t* hbuf  = (half_t*)alloc((size_t)2 * DIM_B * DIM_H * 2);
  half_t* gx16  = (half_t*)alloc((size_t)NROWS * DIM_G * 2);
  half_t* ysA   = (half_t*)alloc((size_t)NROWS * DIM_H * 2);
  half_t* ysB   = (half_t*)alloc((size_t)NROWS * DIM_H * 2);
  if (off > ws_size) return;  // workspace too small; nothing sane to do

  auto cvt = [&](const float* src, half_t* dst, long long n) {
    f32_to_f16_kernel<<<(unsigned)((n + 255) / 256), 256, 0, stream>>>(src, dst,
                                                                       (int)n);
  };
  auto gemm = [&](const half_t* A, const half_t* W, const float* bias,
                  half_t* o16, float* o32, long long Nrows, int M, int K) {
    long long waves = (Nrows / 16) * (M / 64);
    unsigned blocks = (unsigned)((waves + 7) / 8);
    gemm_xw_kernel<<<blocks, 256, 0, stream>>>(A, W, bias, o16, o32,
                                               (int)Nrows, M, K);
  };
  auto rec = [&](const half_t* gx, unsigned long long tstride,
                 const half_t* U, const float* bhh, half_t* ys, int which) {
    gru_rec_kernel<<<REC_WGS, 1024, REC_LDS_BYTES, stream>>>(
        gx, tstride, U, bhh, ys, hbuf, counters + which, DIM_T);
  };

  zero_u32_kernel<<<1, 64, 0, stream>>>(counters, 4);

  // f32 -> f16 weight / input conversion
  cvt(x,     x16,    NROWS * DIM_IN);
  cvt(eWih0, eW0_16, (long long)DIM_G * DIM_IN);
  cvt(eWhh0, eU0_16, (long long)DIM_G * DIM_H);
  cvt(eWih1, eW1_16, (long long)DIM_G * DIM_H);
  cvt(eWhh1, eU1_16, (long long)DIM_G * DIM_H);
  cvt(elW,   elW16,  (long long)DIM_E * DIM_H);
  cvt(dWih0, dW0_16, (long long)DIM_G * DIM_E);
  cvt(dWhh0, dU0_16, (long long)DIM_G * DIM_H);
  cvt(dWih1, dW1_16, (long long)DIM_G * DIM_H);
  cvt(dWhh1, dU1_16, (long long)DIM_G * DIM_H);
  cvt(dlW,   dlW16,  (long long)DIM_IN * DIM_H);

  // ---- encoder ----
  gemm(x16, eW0_16, eBih0, gx16, nullptr, NROWS, DIM_G, DIM_IN);
  rec(gx16, (unsigned long long)DIM_B * DIM_G, eU0_16, eBhh0, ysA, 0);
  gemm(ysA, eW1_16, eBih1, gx16, nullptr, NROWS, DIM_G, DIM_H);
  rec(gx16, (unsigned long long)DIM_B * DIM_G, eU1_16, eBhh1, ysB, 1);

  // encoder linear: only the last timestep is ever used.
  gemm(ysB + (size_t)(DIM_T - 1) * DIM_B * DIM_H, elW16, elB, z16, nullptr,
       DIM_B, DIM_E, DIM_H);

  // ---- decoder ----
  // z_rep is constant over t: one [64 x 1536] input GEMM, recurrence reads it
  // with t-stride 0.
  gemm(z16, dW0_16, dBih0, gx16, nullptr, DIM_B, DIM_G, DIM_E);
  rec(gx16, 0ull, dU0_16, dBhh0, ysA, 2);
  gemm(ysA, dW1_16, dBih1, gx16, nullptr, NROWS, DIM_G, DIM_H);
  rec(gx16, (unsigned long long)DIM_B * DIM_G, dU1_16, dBhh1, ysB, 3);

  // final linear, f32 output
  gemm(ysB, dlW16, dlB, nullptr, out, NROWS, DIM_IN, DIM_H);
}
